// ImprovedMockMamba_55886114455569
// MI455X (gfx1250) — matrix operand order repaired and checked by reading, not compiled, not run on hardware
//
#include <hip/hip_runtime.h>
#include <math.h>

#define D_MODEL 1024
#define D_STATE 16
#define D_CONV  4
#define D_INNER 2048
#define DT_RANK 64
#define BATCH   4
#define SEQ     2048
#define NROWS   (BATCH * SEQ)           /* 8192 */
#define NPROJ   (DT_RANK + 2 * D_STATE) /* 96 */
#define LN_EPS  1e-5f

typedef __attribute__((ext_vector_type(16))) __bf16 v16bf;
typedef __attribute__((ext_vector_type(8)))  float  v8f;
typedef __attribute__((ext_vector_type(4)))  int    v4i_t;

union FragBF { unsigned u[8]; v16bf v; };

#if __has_builtin(__builtin_amdgcn_global_load_async_to_lds_b128) && \
    __has_builtin(__builtin_amdgcn_s_wait_asynccnt)
#define USE_ASYNC_LDS 1
#else
#define USE_ASYNC_LDS 0
#endif

#define AS_GLOBAL __attribute__((address_space(1)))
#define AS_LDS    __attribute__((address_space(3)))

__device__ __forceinline__ unsigned bf16_rne(float f) {
  unsigned u = __float_as_uint(f);
  u += 0x7FFFu + ((u >> 16) & 1u);
  return u >> 16;
}
__device__ __forceinline__ float bf16u_to_f(unsigned h) {
  return __uint_as_float(h << 16);
}
// Split two f32 into (hi, lo) bf16 pairs, first element in [15:0].
__device__ __forceinline__ void split_pack2(float f0, float f1,
                                            unsigned &hi, unsigned &lo) {
  unsigned h0 = bf16_rne(f0), h1 = bf16_rne(f1);
  float r0 = f0 - bf16u_to_f(h0);
  float r1 = f1 - bf16u_to_f(h1);
  unsigned l0 = bf16_rne(r0), l1 = bf16_rne(r1);
  hi = h0 | (h1 << 16);
  lo = l0 | (l1 << 16);
}
// Reconstruct f32 from packed hi/lo pair word, half = 0 or 1.
__device__ __forceinline__ float unpack_bf16x2(unsigned h, unsigned l, int half) {
  unsigned hb = (h >> (16 * half)) & 0xFFFFu;
  unsigned lb = (l >> (16 * half)) & 0xFFFFu;
  return bf16u_to_f(hb) + bf16u_to_f(lb);
}

// 16-byte global -> LDS copy: async on CDNA5 toolchains, plain copy otherwise.
__device__ __forceinline__ void copy16_g2l(const unsigned* g, unsigned* l) {
#if USE_ASYNC_LDS
  __builtin_amdgcn_global_load_async_to_lds_b128(
      (AS_GLOBAL v4i_t*)g, (AS_LDS v4i_t*)l, 0, 0);
#else
  *(uint4*)l = *(const uint4*)g;
#endif
}
__device__ __forceinline__ void wait_staging() {
#if USE_ASYNC_LDS
  __builtin_amdgcn_s_wait_asynccnt(0);
#endif
}

// ---------------------------------------------------------------------------
// Pre-packed bf16x2 GEMM:  C = act(A @ B + bias)
//   Ah/Al: [M][K/2] dwords (bf16 pair along K, packed from f32 A)
//   Bh/Bl: [N][K/2] dwords (transpose-packed from f32 B[K][N])
// Block: 256 thr = 8 waves; tile 32(M) x 128(N); wave tile 16x32 (2 acc).
// K step 32; double-buffered LDS; async global->LDS staging.
// ACT: 0 = none, 1 = softplus.
// ---------------------------------------------------------------------------
template <int ACT>
__global__ __launch_bounds__(256)
void gemm_bf16pk_wmma(const unsigned* __restrict__ Ah_g, const unsigned* __restrict__ Al_g,
                      const unsigned* __restrict__ Bh_g, const unsigned* __restrict__ Bl_g,
                      const float* __restrict__ bias, float* __restrict__ C,
                      int M, int N, int K) {
  // stride 20 dwords: rows stay 16B-aligned for b128, 20*m mod 64 distinct
  __shared__ unsigned Ah[2][32][20], Al[2][32][20];
  __shared__ unsigned Bh[2][128][20], Bl[2][128][20];

  const int tid  = threadIdx.x;
  const int lane = tid & 31;
  const int wid  = tid >> 5;
  const int lm = lane & 15, lh = lane >> 4;
  const int wm = wid >> 2, wn = wid & 3;
  const int blockM = blockIdx.y * 32;
  const int blockN = blockIdx.x * 128;
  const int KPd = K >> 1;

  v8f acc0 = {}, acc1 = {};

  // 1280 b128 chunks per tile: A(h,l): 2*128, B(h,l): 2*512 -> 5 per thread
  auto stage = [&](int buf, int kp0) {
#pragma unroll
    for (int i = 0; i < 5; ++i) {
      int o = tid + i * 256;
      if (o < 256) {
        int arr = o >> 7, idx = o & 127;
        int m = idx >> 2, c = idx & 3;
        const unsigned* g =
            (arr ? Al_g : Ah_g) + (size_t)(blockM + m) * KPd + kp0 + c * 4;
        unsigned* lp = arr ? &Al[buf][m][c * 4] : &Ah[buf][m][c * 4];
        copy16_g2l(g, lp);
      } else {
        int o2 = o - 256;
        int arr = o2 >> 9, idx = o2 & 511;
        int n = idx >> 2, c = idx & 3;
        if (blockN + n < N) {  // OOB cols never stored (column-isolated)
          const unsigned* g =
              (arr ? Bl_g : Bh_g) + (size_t)(blockN + n) * KPd + kp0 + c * 4;
          unsigned* lp = arr ? &Bl[buf][n][c * 4] : &Bh[buf][n][c * 4];
          copy16_g2l(g, lp);
        }
      }
    }
  };

  const int T = K >> 5;
  stage(0, 0);
  for (int t = 0; t < T; ++t) {
    wait_staging();
    __syncthreads();
    if (t + 1 < T) stage((t + 1) & 1, (t + 1) << 4);

    const int buf = t & 1;
    FragBF ah, al, bh0, bl0, bh1, bl1;
#pragma unroll
    for (int v = 0; v < 8; ++v) {
      // A (16-bit 16x32): VGPR v<4 -> K=2v (+8 if lane>=16); v>=4 -> K=16+2(v-4)(+8)
      int akp = (v < 4) ? (v + 4 * lh) : (4 + v + 4 * lh);
      ah.u[v] = Ah[buf][wm * 16 + lm][akp];
      al.u[v] = Al[buf][wm * 16 + lm][akp];
      // B (16-bit 32x16): same K order as A
      int bkp = (v < 4) ? (v + 4 * lh) : (4 + v + 4 * lh);
      bh0.u[v] = Bh[buf][wn * 32 + lm][bkp];
      bl0.u[v] = Bl[buf][wn * 32 + lm][bkp];
      bh1.u[v] = Bh[buf][wn * 32 + 16 + lm][bkp];
      bl1.u[v] = Bl[buf][wn * 32 + 16 + lm][bkp];
    }
    acc0 = __builtin_amdgcn_wmma_f32_16x16x32_bf16(false, ah.v, false, bh0.v, (short)0, acc0, false, false);
    acc0 = __builtin_amdgcn_wmma_f32_16x16x32_bf16(false, ah.v, false, bl0.v, (short)0, acc0, false, false);
    acc0 = __builtin_amdgcn_wmma_f32_16x16x32_bf16(false, al.v, false, bh0.v, (short)0, acc0, false, false);
    acc1 = __builtin_amdgcn_wmma_f32_16x16x32_bf16(false, ah.v, false, bh1.v, (short)0, acc1, false, false);
    acc1 = __builtin_amdgcn_wmma_f32_16x16x32_bf16(false, ah.v, false, bl1.v, (short)0, acc1, false, false);
    acc1 = __builtin_amdgcn_wmma_f32_16x16x32_bf16(false, al.v, false, bh1.v, (short)0, acc1, false, false);
    __syncthreads();
  }

  // epilogue: C/D layout row = r + 8*(lane>=16)
#pragma unroll
  for (int half = 0; half < 2; ++half) {
    int col = blockN + wn * 32 + half * 16 + lm;
    if (col < N) {
      float bv = bias ? bias[col] : 0.f;
      v8f a = half ? acc1 : acc0;
#pragma unroll
      for (int r = 0; r < 8; ++r) {
        int row = blockM + wm * 16 + r + 8 * lh;
        if (row < M) {
          float v = a[r] + bv;
          if (ACT == 1) v = (v > 20.f) ? v : log1pf(__expf(v));
          C[(size_t)row * N + col] = v;
        }
      }
    }
  }
}

// ---------------------------------------------------------------------------
// Pack activations [rows][K] f32 -> hi/lo bf16 pair dwords (pair along K).
// ---------------------------------------------------------------------------
__global__ __launch_bounds__(256)
void pack_pairs_kernel(const float* __restrict__ src, unsigned* __restrict__ h,
                       unsigned* __restrict__ l, int npairs) {
  int i = blockIdx.x * 256 + threadIdx.x;
  if (i >= npairs) return;
  float2 f = ((const float2*)src)[i];
  split_pack2(f.x, f.y, h[i], l[i]);
}

// ---------------------------------------------------------------------------
// Transpose-pack weights: B[K][N] f32 -> Bh/Bl [N][K/2] dwords. LDS-tiled.
// tile: 64 K-rows (32 pairs) x 64 N-cols. grid = (ceil(N/64), K/64)
// ---------------------------------------------------------------------------
__global__ __launch_bounds__(256)
void pack_B_kernel(const float* __restrict__ B, unsigned* __restrict__ Bh,
                   unsigned* __restrict__ Bl, int K, int N) {
  __shared__ unsigned sh[64][33], sl[64][33];
  const int n0 = blockIdx.x * 64;
  const int k0 = blockIdx.y * 64;
  const int t = threadIdx.x;
  const int KPd = K >> 1;
#pragma unroll
  for (int i = 0; i < 8; ++i) {   // coalesced reads over n
    int e = t + i * 256;
    int kp = e >> 6, n = e & 63;
    int gk = k0 + 2 * kp, gn = n0 + n;
    float f0 = 0.f, f1 = 0.f;
    if (gn < N) {
      f0 = B[(size_t)gk * N + gn];
      f1 = B[(size_t)(gk + 1) * N + gn];
    }
    split_pack2(f0, f1, sh[n][kp], sl[n][kp]);
  }
  __syncthreads();
#pragma unroll
  for (int i = 0; i < 8; ++i) {   // coalesced writes over kp
    int e = t + i * 256;
    int n = e >> 5, kp = e & 31;
    int gn = n0 + n;
    if (gn < N) {
      Bh[(size_t)gn * KPd + (k0 >> 1) + kp] = sh[n][kp];
      Bl[(size_t)gn * KPd + (k0 >> 1) + kp] = sl[n][kp];
    }
  }
}

// ---------------------------------------------------------------------------
// u = silu(causal depthwise conv1d(xc) + conv_b); emits packed bf16x2 only.
// One thread per channel pair (d, d+1).
// ---------------------------------------------------------------------------
__global__ __launch_bounds__(256)
void conv_silu_pack_kernel(const float* __restrict__ xz,
                           const float* __restrict__ conv_w,
                           const float* __restrict__ conv_b,
                           unsigned* __restrict__ uh, unsigned* __restrict__ ul) {
  int i = blockIdx.x * 256 + threadIdx.x;
  if (i >= NROWS * (D_INNER / 2)) return;
  int dp = i & (D_INNER / 2 - 1);
  int bl = i >> 10;
  int l = bl & (SEQ - 1);
  int d0 = dp * 2;
  float acc0 = conv_b[d0], acc1 = conv_b[d0 + 1];
#pragma unroll
  for (int k = 0; k < D_CONV; ++k) {
    int ls = l + k - (D_CONV - 1);
    if (ls >= 0) {
      const float2 v =
          *(const float2*)&xz[(size_t)(bl + k - (D_CONV - 1)) * (2 * D_INNER) + d0];
      acc0 = fmaf(v.x, conv_w[d0 * D_CONV + k], acc0);
      acc1 = fmaf(v.y, conv_w[(d0 + 1) * D_CONV + k], acc1);
    }
  }
  acc0 = acc0 / (1.f + __expf(-acc0));
  acc1 = acc1 / (1.f + __expf(-acc1));
  split_pack2(acc0, acc1, uh[i], ul[i]);
}

// ---------------------------------------------------------------------------
// LN(64) on dt (emits packed bf16x2), silu(LN(16)) on B (lanes 0-15) and
// C (lanes 16-31). One wave per row.
// ---------------------------------------------------------------------------
__global__ __launch_bounds__(128)
void ln_proj_kernel(const float* __restrict__ xproj,
                    const float* __restrict__ g_dt, const float* __restrict__ b_dt,
                    const float* __restrict__ g_B,  const float* __restrict__ b_B,
                    const float* __restrict__ g_C,  const float* __restrict__ b_C,
                    unsigned* __restrict__ dth, unsigned* __restrict__ dtl,
                    float* __restrict__ Bmat, float* __restrict__ Cmat) {
  int lane = threadIdx.x & 31;
  int row  = blockIdx.x * 4 + (threadIdx.x >> 5);
  const float* xp = xproj + (size_t)row * NPROJ;

  // LN over 64: lane handles elements 2*lane, 2*lane+1 (pair-friendly)
  float2 v = ((const float2*)xp)[lane];
  float sum = v.x + v.y;
#pragma unroll
  for (int m = 16; m >= 1; m >>= 1) sum += __shfl_xor(sum, m, 32);
  float mu = sum * (1.f / 64.f);
  float d0 = v.x - mu, d1 = v.y - mu;
  float sq = d0 * d0 + d1 * d1;
#pragma unroll
  for (int m = 16; m >= 1; m >>= 1) sq += __shfl_xor(sq, m, 32);
  float inv = rsqrtf(sq * (1.f / 64.f) + LN_EPS);
  float y0 = d0 * inv * g_dt[2 * lane]     + b_dt[2 * lane];
  float y1 = d1 * inv * g_dt[2 * lane + 1] + b_dt[2 * lane + 1];
  split_pack2(y0, y1, dth[(size_t)row * (DT_RANK / 2) + lane],
              dtl[(size_t)row * (DT_RANK / 2) + lane]);

  // LN over 16 + silu
  int s   = lane & 15;
  int isC = lane >> 4;
  float x = xp[DT_RANK + isC * D_STATE + s];
  float sm = x;
#pragma unroll
  for (int m = 8; m >= 1; m >>= 1) sm += __shfl_xor(sm, m, 16);
  float mu2 = sm * (1.f / 16.f);
  float dd = x - mu2;
  float s2 = dd * dd;
#pragma unroll
  for (int m = 8; m >= 1; m >>= 1) s2 += __shfl_xor(s2, m, 16);
  float inv2 = rsqrtf(s2 * (1.f / 16.f) + LN_EPS);
  float g  = isC ? g_C[s] : g_B[s];
  float bb = isC ? b_C[s] : b_B[s];
  float y = dd * inv2 * g + bb;
  y = y / (1.f + __expf(-y));
  (isC ? Cmat : Bmat)[(size_t)row * D_STATE + s] = y;
}

// ---------------------------------------------------------------------------
// Selective scan + D-skip + silu(z) gate; emits packed bf16x2 yg.
// Thread = (d, s); 16 lanes per channel; wave holds channel pair (d even, d+1).
// ---------------------------------------------------------------------------
__global__ __launch_bounds__(256)
void scan_kernel(const float* __restrict__ delta,
                 const unsigned* __restrict__ uh, const unsigned* __restrict__ ul,
                 const float* __restrict__ Bmat, const float* __restrict__ Cmat,
                 const float* __restrict__ xz, const float* __restrict__ A_log,
                 const float* __restrict__ Dskip,
                 unsigned* __restrict__ ygh, unsigned* __restrict__ ygl) {
  int tid = threadIdx.x;
  int s  = tid & 15, dl = tid >> 4;
  int d  = blockIdx.x * 16 + dl;
  int b  = blockIdx.y;
  float Aval = __expf(A_log[d * D_STATE + s]); // A = exp(A_log)
  float Dv   = Dskip[d];
  float h = 0.f;
  size_t rowBase = (size_t)b * SEQ;
  for (int l = 0; l < SEQ; ++l) {
    size_t r = rowBase + l;
    float delta_t = delta[r * D_INNER + d];
    float u_t = unpack_bf16x2(uh[r * (D_INNER / 2) + (d >> 1)],
                              ul[r * (D_INNER / 2) + (d >> 1)], d & 1);
    float Bv = Bmat[r * D_STATE + s];
    float Cv = Cmat[r * D_STATE + s];
    float dA = __expf(delta_t * Aval);
    h = fmaf(dA, h, delta_t * u_t * Bv);
    float p = h * Cv;
    p += __shfl_xor(p, 8, 16);
    p += __shfl_xor(p, 4, 16);
    p += __shfl_xor(p, 2, 16);
    p += __shfl_xor(p, 1, 16);
    float zv = xz[r * (2 * D_INNER) + D_INNER + d];
    float sz = zv / (1.f + __expf(-zv));
    float yfull = (p + Dv * u_t) * sz;          // valid on every lane
    float ypartner = __shfl_xor(yfull, 16, 32); // other channel of the wave
    if ((tid & 31) == 0) {                       // d even; partner = d+1
      unsigned hw, lw;
      split_pack2(yfull, ypartner, hw, lw);
      ygh[r * (D_INNER / 2) + (d >> 1)] = hw;
      ygl[r * (D_INNER / 2) + (d >> 1)] = lw;
    }
  }
}

// ---------------------------------------------------------------------------
extern "C" void kernel_launch(void* const* d_in, const int* in_sizes, int n_in,
                              void* d_out, int out_size, void* d_ws, size_t ws_size,
                              hipStream_t stream) {
  (void)in_sizes; (void)n_in; (void)out_size; (void)ws_size;
  const float* x      = (const float*)d_in[0];
  const float* Wi     = (const float*)d_in[1];
  const float* bi     = (const float*)d_in[2];
  const float* conv_w = (const float*)d_in[3];
  const float* conv_b = (const float*)d_in[4];
  const float* Wx     = (const float*)d_in[5];
  const float* Wdt    = (const float*)d_in[6];
  const float* bdt    = (const float*)d_in[7];
  const float* A_log  = (const float*)d_in[8];
  const float* Dskip  = (const float*)d_in[9];
  const float* Wo     = (const float*)d_in[10];
  const float* bo     = (const float*)d_in[11];
  const float* g_dt   = (const float*)d_in[12];
  const float* b_dt   = (const float*)d_in[13];
  const float* g_B    = (const float*)d_in[14];
  const float* b_B    = (const float*)d_in[15];
  const float* g_C    = (const float*)d_in[16];
  const float* b_C    = (const float*)d_in[17];

  unsigned* w = (unsigned*)d_ws;
  float*    xz    = (float*)w;     w += (size_t)NROWS * 2 * D_INNER;   // f32
  float*    delta = (float*)w;     w += (size_t)NROWS * D_INNER;       // f32
  float*    xproj = (float*)w;     w += (size_t)NROWS * NPROJ;         // f32
  float*    Bm    = (float*)w;     w += (size_t)NROWS * D_STATE;       // f32
  float*    Cm    = (float*)w;     w += (size_t)NROWS * D_STATE;       // f32
  unsigned* xh    = w;             w += (size_t)NROWS * (D_MODEL / 2);
  unsigned* xl    = w;             w += (size_t)NROWS * (D_MODEL / 2);
  unsigned* uh    = w;             w += (size_t)NROWS * (D_INNER / 2);
  unsigned* ul    = w;             w += (size_t)NROWS * (D_INNER / 2);
  unsigned* dth   = w;             w += (size_t)NROWS * (DT_RANK / 2);
  unsigned* dtl   = w;             w += (size_t)NROWS * (DT_RANK / 2);
  unsigned* ygh   = w;             w += (size_t)NROWS * (D_INNER / 2);
  unsigned* ygl   = w;             w += (size_t)NROWS * (D_INNER / 2);
  unsigned* Wih   = w;             w += (size_t)(2 * D_INNER) * (D_MODEL / 2);
  unsigned* Wil   = w;             w += (size_t)(2 * D_INNER) * (D_MODEL / 2);
  unsigned* Wxh   = w;             w += (size_t)NPROJ * (D_INNER / 2);
  unsigned* Wxl   = w;             w += (size_t)NPROJ * (D_INNER / 2);
  unsigned* Wdth  = w;             w += (size_t)D_INNER * (DT_RANK / 2);
  unsigned* Wdtl  = w;             w += (size_t)D_INNER * (DT_RANK / 2);
  unsigned* Woh   = w;             w += (size_t)D_MODEL * (D_INNER / 2);
  unsigned* Wol   = w;             w += (size_t)D_MODEL * (D_INNER / 2);

  // --- one-time operand packing (weights + x) ---
  {
    int np = NROWS * (D_MODEL / 2);
    pack_pairs_kernel<<<(np + 255) / 256, 256, 0, stream>>>(x, xh, xl, np);
  }
  pack_B_kernel<<<dim3((2 * D_INNER) / 64, D_MODEL / 64), 256, 0, stream>>>(Wi, Wih, Wil, D_MODEL, 2 * D_INNER);
  pack_B_kernel<<<dim3((NPROJ + 63) / 64, D_INNER / 64), 256, 0, stream>>>(Wx, Wxh, Wxl, D_INNER, NPROJ);
  pack_B_kernel<<<dim3(D_INNER / 64, DT_RANK / 64), 256, 0, stream>>>(Wdt, Wdth, Wdtl, DT_RANK, D_INNER);
  pack_B_kernel<<<dim3(D_MODEL / 64, D_INNER / 64), 256, 0, stream>>>(Wo, Woh, Wol, D_INNER, D_MODEL);

  // 1. xz = x @ Wi + bi
  gemm_bf16pk_wmma<0><<<dim3((2 * D_INNER) / 128, NROWS / 32), 256, 0, stream>>>(
      xh, xl, Wih, Wil, bi, xz, NROWS, 2 * D_INNER, D_MODEL);
  // 2. u = silu(depthwise causal conv(xc))  (packed bf16x2)
  conv_silu_pack_kernel<<<(NROWS * (D_INNER / 2)) / 256, 256, 0, stream>>>(
      xz, conv_w, conv_b, uh, ul);
  // 3. xproj = u @ Wx (no bias)
  gemm_bf16pk_wmma<0><<<dim3((NPROJ + 127) / 128, NROWS / 32), 256, 0, stream>>>(
      uh, ul, Wxh, Wxl, nullptr, xproj, NROWS, NPROJ, D_INNER);
  // 4. layernorms (+ packed dt_lo)
  ln_proj_kernel<<<NROWS / 4, 128, 0, stream>>>(xproj, g_dt, b_dt, g_B, b_B,
                                                g_C, b_C, dth, dtl, Bm, Cm);
  // 5. delta = softplus(dt_lo @ Wdt + bdt)
  gemm_bf16pk_wmma<1><<<dim3(D_INNER / 128, NROWS / 32), 256, 0, stream>>>(
      dth, dtl, Wdth, Wdtl, bdt, delta, NROWS, D_INNER, DT_RANK);
  // 6. selective scan + D-skip + silu(z) gate (packed yg)
  scan_kernel<<<dim3(D_INNER / 16, BATCH), 256, 0, stream>>>(
      delta, uh, ul, Bm, Cm, xz, A_log, Dskip, ygh, ygl);
  // 7. out = yg @ Wo + bo
  gemm_bf16pk_wmma<0><<<dim3(D_MODEL / 128, NROWS / 32), 256, 0, stream>>>(
      ygh, ygl, Woh, Wol, bo, (float*)d_out, NROWS, D_MODEL, D_INNER);
}
